// QuantizedLinear_79499844649451
// MI455X (gfx1250) — matrix operand order repaired and checked by reading, not compiled, not run on hardware
//
#include <hip/hip_runtime.h>
#include <hip/hip_bf16.h>

#define IN_F  4096
#define OUT_F 4096
#define BM 128
#define BN 256
#define BK 32
#define LDA 40   // padded LDS row stride in elements (80 B = 5 x 16 B, b128-aligned)
#define LDB 40

typedef __attribute__((ext_vector_type(16))) __bf16 v16bf;
typedef __attribute__((ext_vector_type(8)))  float  v8f;
typedef int v4i_b128 __attribute__((vector_size(16)));   // matches builtin param type
typedef __attribute__((address_space(1))) v4i_b128* gas_b128;
typedef __attribute__((address_space(3))) v4i_b128* las_b128;

union FragAB { uint4 u4[2]; v16bf v; };

static __device__ __forceinline__ unsigned short f2bf_raw(float f) {
    __bf16 h = (__bf16)f;
    return __builtin_bit_cast(unsigned short, h);
}

// ---- CDNA5 async global->LDS copy (ASYNCcnt path), with safe fallback -----
#if defined(__HIP_DEVICE_COMPILE__) && __has_builtin(__builtin_amdgcn_global_load_async_to_lds_b128)
#define HAVE_ASYNC_LDS 1
#else
#define HAVE_ASYNC_LDS 0
#endif

static __device__ __forceinline__ void cp_b128(const unsigned short* g, unsigned short* l) {
#if HAVE_ASYNC_LDS
    __builtin_amdgcn_global_load_async_to_lds_b128(
        (gas_b128)(unsigned long long)g,
        (las_b128)(unsigned int)(unsigned long long)l,
        0, 0);
#else
    *(uint4*)l = *(const uint4*)g;
#endif
}

static __device__ __forceinline__ void wait_async_copies() {
#if HAVE_ASYNC_LDS
#if __has_builtin(__builtin_amdgcn_s_wait_asynccnt)
    __builtin_amdgcn_s_wait_asynccnt(0);
#else
    asm volatile("s_wait_asynccnt 0" ::: "memory");
#endif
#endif
}

// ---- scheduler shaping: 0x100 = DS read, 0x008 = MFMA/WMMA ----------------
#if defined(__HIP_DEVICE_COMPILE__) && __has_builtin(__builtin_amdgcn_sched_group_barrier)
#define SCHED_DS(n)   __builtin_amdgcn_sched_group_barrier(0x100, n, 0)
#define SCHED_WMMA(n) __builtin_amdgcn_sched_group_barrier(0x008, n, 0)
#else
#define SCHED_DS(n)
#define SCHED_WMMA(n)
#endif

// ---------------------------------------------------------------------------
// Kernel 1: trellis decode -> bf16 W stored row-major [out_f][in_f] (N x K)
// ---------------------------------------------------------------------------
__global__ __launch_bounds__(256) void k_dequant(const int* __restrict__ trellis,
                                                 const float* __restrict__ tlut,
                                                 unsigned short* __restrict__ Wbf) {
    __shared__ int w[2][32];
    const int tid  = threadIdx.x;
    const int sub  = tid >> 7;
    const int s    = tid & 127;
    const int tile = blockIdx.x * 2 + sub;

    if (s < 32) w[sub][s] = trellis[tile * 32 + s] & 0xFFFF;
    __syncthreads();

    const int* wd = w[sub];
#define NIB(i) ((wd[(i) >> 2] >> (12 - 4 * ((i) & 3))) & 0xF)
    const int i1 = (s - 1) & 127, i2 = (s - 2) & 127, i3 = (s - 3) & 127;
    const int state = NIB(s) | (NIB(i1) << 4) | (NIB(i2) << 8) | (NIB(i3) << 12);
#undef NIB

    const float v0 = tlut[state * 2 + 0];
    const float v1 = tlut[state * 2 + 1];

    const int e   = s * 2;
    const int row = (tile >> 8) * 16 + (e >> 4);
    const int col = (tile & 255) * 16 + (e & 15);
    const unsigned int packed =
        (unsigned int)f2bf_raw(v0) | ((unsigned int)f2bf_raw(v1) << 16);
    *(unsigned int*)&Wbf[(size_t)row * IN_F + col] = packed;
}

// ---------------------------------------------------------------------------
// Kernel 2: X1 = bf16( fwht(x * SU) ).
// ---------------------------------------------------------------------------
__global__ __launch_bounds__(256) void k_fwht_in(const float* __restrict__ x,
                                                 const float* __restrict__ SU,
                                                 unsigned short* __restrict__ X1) {
    __shared__ float buf[IN_F];
    const int row = blockIdx.x;
    const int tid = threadIdx.x;
    const float* xr = x + (size_t)row * IN_F;

#pragma unroll
    for (int j = 0; j < 16; ++j) {
        const int i = tid + j * 256;
        buf[i] = xr[i] * SU[i];
    }
    __syncthreads();

    for (int b = 0; b < 12; ++b) {
        const int h = 1 << b;
#pragma unroll
        for (int pp = 0; pp < 8; ++pp) {
            const int p = tid + pp * 256;
            const int i = ((p >> b) << (b + 1)) | (p & (h - 1));
            const float a = buf[i], c = buf[i + h];
            buf[i]     = a + c;
            buf[i + h] = a - c;
        }
        __syncthreads();
    }

    unsigned short* o = X1 + (size_t)row * IN_F;
#pragma unroll
    for (int j = 0; j < 16; ++j) {
        const int i = tid + j * 256;
        o[i] = f2bf_raw(buf[i] * 0.015625f);
    }
}

// ---------------------------------------------------------------------------
// Kernel 3: Y = X1 @ W^T.  Block 128x256, BK=32, double-buffered LDS with
// async global->LDS fills. 8 waves, each computing 64x64 = 16 WMMAs per
// k-slice against 16 ds_load_b128.  Single (non-unrolled) loop body keeps
// one phi per accumulator so the RA can pin them in place.
// ---------------------------------------------------------------------------
__global__ __launch_bounds__(256) void k_gemm(const unsigned short* __restrict__ X1,
                                              const unsigned short* __restrict__ Wbf,
                                              float* __restrict__ Y) {
    __shared__ __align__(16) unsigned short Asm[2][BM * LDA];
    __shared__ __align__(16) unsigned short Bsm[2][BN * LDB];

    const int tid  = threadIdx.x;
    const int lane = tid & 31;
    const int wid  = tid >> 5;
    const int wm   = wid >> 2;     // 0..1 -> 64-row slice
    const int wn   = wid & 3;      // 0..3 -> 64-col slice
    const int l16  = lane & 15;
    const int hi   = lane >> 4;
    const int m0   = blockIdx.y * BM;
    const int n0   = blockIdx.x * BN;

    v8f acc[4][4];
    const v8f vzero = {0.f, 0.f, 0.f, 0.f, 0.f, 0.f, 0.f, 0.f};
#pragma unroll
    for (int a = 0; a < 4; ++a)
#pragma unroll
        for (int b = 0; b < 4; ++b) acc[a][b] = vzero;

    // Stage one BK slice of A (128x32, 512 x b128) and B (256x32, 1024 x b128)
    auto stage = [&](int buf, int kk) {
#pragma unroll
        for (int j = 0; j < 2; ++j) {           // A: 2 chunks / thread
            const int q = tid + j * 256;
            const int r = q >> 2, ch = (q & 3) * 8;
            cp_b128(&X1[(size_t)(m0 + r) * IN_F + kk + ch], &Asm[buf][r * LDA + ch]);
        }
#pragma unroll
        for (int j = 0; j < 4; ++j) {           // B: 4 chunks / thread
            const int q = tid + j * 256;
            const int r = q >> 2, ch = (q & 3) * 8;
            cp_b128(&Wbf[(size_t)(n0 + r) * IN_F + kk + ch], &Bsm[buf][r * LDB + ch]);
        }
    };

    auto compute = [&](int buf) {
        // B col N = l16; lane<16: K0..15; lane>=16: K16..31  (persistent)
        FragAB bfr[4];
#pragma unroll
        for (int nt = 0; nt < 4; ++nt) {
            const unsigned short* p = &Bsm[buf][(wn * 64 + nt * 16 + l16) * LDB + hi * 16];
            bfr[nt].u4[0] = *(const uint4*)p;
            bfr[nt].u4[1] = *(const uint4*)(p + 8);
        }
        // A lane<16: K0..15; lane>=16: K16..31  (streamed)
#pragma unroll
        for (int mt = 0; mt < 4; ++mt) {
            FragAB a;
            const unsigned short* p = &Asm[buf][(wm * 64 + mt * 16 + l16) * LDA + hi * 16];
            a.u4[0] = *(const uint4*)p;
            a.u4[1] = *(const uint4*)(p + 8);
#pragma unroll
            for (int nt = 0; nt < 4; ++nt)
                acc[mt][nt] = __builtin_amdgcn_wmma_f32_16x16x32_bf16(
                    false, a.v, false, bfr[nt].v,
                    (short)0, acc[mt][nt], false, false);
        }
        // Shape: 8 B-frag DS reads, then 4x (2 DS reads, 4 WMMA)
        SCHED_DS(8);
        SCHED_DS(2); SCHED_WMMA(4);
        SCHED_DS(2); SCHED_WMMA(4);
        SCHED_DS(2); SCHED_WMMA(4);
        SCHED_DS(2); SCHED_WMMA(4);
    };

    stage(0, 0);
    wait_async_copies();
    __syncthreads();

    int buf = 0;
#pragma clang loop unroll(disable)
    for (int kk = BK; kk < IN_F; kk += BK) {
        stage(buf ^ 1, kk);      // fill next buffer (async overlaps WMMAs)
        compute(buf);
        wait_async_copies();
        __syncthreads();
        buf ^= 1;
    }
    compute(buf);

    // Store: lane<16 -> M=j, lane>=16 -> M=j+8; N = l16
#pragma unroll
    for (int mt = 0; mt < 4; ++mt)
#pragma unroll
        for (int nt = 0; nt < 4; ++nt) {
            const int col = n0 + wn * 64 + nt * 16 + l16;
            const int rb  = m0 + wm * 64 + mt * 16 + hi * 8;
#pragma unroll
            for (int j = 0; j < 8; ++j)
                Y[(size_t)(rb + j) * OUT_F + col] = acc[mt][nt][j];
        }
}

// ---------------------------------------------------------------------------
// Kernel 4: in-place row FWHT on f32 output, then * SV * (1/64).
// ---------------------------------------------------------------------------
__global__ __launch_bounds__(256) void k_fwht_out(float* __restrict__ y,
                                                  const float* __restrict__ SV) {
    __shared__ float buf[OUT_F];
    const int row = blockIdx.x;
    const int tid = threadIdx.x;
    float* yr = y + (size_t)row * OUT_F;

#pragma unroll
    for (int j = 0; j < 16; ++j) {
        const int i = tid + j * 256;
        buf[i] = yr[i];
    }
    __syncthreads();

    for (int b = 0; b < 12; ++b) {
        const int h = 1 << b;
#pragma unroll
        for (int pp = 0; pp < 8; ++pp) {
            const int p = tid + pp * 256;
            const int i = ((p >> b) << (b + 1)) | (p & (h - 1));
            const float a = buf[i], c = buf[i + h];
            buf[i]     = a + c;
            buf[i + h] = a - c;
        }
        __syncthreads();
    }

#pragma unroll
    for (int j = 0; j < 16; ++j) {
        const int i = tid + j * 256;
        yr[i] = buf[i] * SV[i] * 0.015625f;
    }
}

// ---------------------------------------------------------------------------
extern "C" void kernel_launch(void* const* d_in, const int* in_sizes, int n_in,
                              void* d_out, int out_size, void* d_ws, size_t ws_size,
                              hipStream_t stream) {
    const float* input   = (const float*)d_in[0];
    const int*   trellis = (const int*)  d_in[1];
    const float* tlut    = (const float*)d_in[2];
    const float* SU      = (const float*)d_in[3];
    const float* SV      = (const float*)d_in[4];
    float*       out     = (float*)d_out;

    const int M = in_sizes[0] / IN_F;  // 16384 rows

    unsigned short* Wbf = (unsigned short*)d_ws;
    unsigned short* X1  = (unsigned short*)((char*)d_ws + (size_t)OUT_F * IN_F * 2);

    const int num_tiles = (OUT_F / 16) * (IN_F / 16);
    k_dequant<<<num_tiles / 2, 256, 0, stream>>>(trellis, tlut, Wbf);
    k_fwht_in<<<M, 256, 0, stream>>>(input, SU, X1);
    dim3 grid(OUT_F / BN, M / BM);
    k_gemm<<<grid, 256, 0, stream>>>(X1, Wbf, out);
    k_fwht_out<<<M, 256, 0, stream>>>(out, SV);
}
